// Multi_Head_attention_1606317769372
// MI455X (gfx1250) — compile-verified, matrix operand order checked by reading, not run on hardware
//
#include <hip/hip_runtime.h>

#define Bsz   2
#define Sdim  2048
#define Ddim  512
#define Hdim  512
#define NHEAD 8
#define DH    64
#define NTOK  (Bsz * Sdim)   // 4096

typedef __bf16 bf16_t;
typedef __bf16 v16bf __attribute__((ext_vector_type(16)));
typedef float  v8f   __attribute__((ext_vector_type(8)));

// float -> bf16 round-to-nearest-even via bit manipulation
__device__ __forceinline__ bf16_t f2bf(float f) {
  union { float f; unsigned u; } a; a.f = f;
  unsigned r = a.u + 0x7FFFu + ((a.u >> 16) & 1u);
  unsigned short hs = (unsigned short)(r >> 16);
  union { unsigned short s; bf16_t b; } o; o.s = hs;
  return o.b;
}

// Load one 16x32 bf16 A-fragment (or 32x16 B-fragment from a [col][k] tile).
// ISA layout: lane<16 -> row (lane&15), K = {k0..k0+7, k0+16..k0+23}
//             lane>=16 -> same row,     K = {k0+8..k0+15, k0+24..k0+31}
// ld must be a multiple of 8 so both 16B vector loads stay aligned.
__device__ __forceinline__ v16bf load_frag(const bf16_t* t, int ld, int row0, int k0, int lane) {
  const int half = (lane >> 4) & 1;
  const int r = row0 + (lane & 15);
  const bf16_t* p = t + r * ld + k0 + half * 8;
  union { uint4 u[2]; v16bf v; } u;
  u.u[0] = *(const uint4*)(p);        // K = k0+half*8 .. +7
  u.u[1] = *(const uint4*)(p + 16);   // K = k0+16+half*8 .. +7
  return u.v;
}

__device__ __forceinline__ v8f wmma_bf16(v16bf a, v16bf b, v8f c) {
  return __builtin_amdgcn_wmma_f32_16x16x32_bf16(false, a, false, b, (short)0, c, false, false);
}

// ---------------------------------------------------------------------------
// Tensor Data Mover: 64x64 bf16 2D tile, global row stride Sdim elements,
// into LDS with row stride 72 elements (pad 4 DWORDs after every 32 DWORDs).
// D# layout per CDNA5 ISA.  This toolchain exposes the
// 6-arg builtin: (u32x4 g0, i32x8 g1, i32x4 g2, i32x4 g3, i32x8, i32 cpol).
// ---------------------------------------------------------------------------
#if __has_builtin(__builtin_amdgcn_tensor_load_to_lds)
#define HAVE_TDM 1
typedef unsigned u32x4_t __attribute__((ext_vector_type(4)));
typedef int      i32x4_t __attribute__((ext_vector_type(4)));
typedef int      i32x8_t __attribute__((ext_vector_type(8)));

__device__ __forceinline__ void tdm_load_64x64_pad72(const bf16_t* gsrc, bf16_t* lds_dst) {
  unsigned long long ga = (unsigned long long)(size_t)gsrc;
  unsigned lds_addr = (unsigned)(size_t)lds_dst;  // low 32 bits of generic ptr == LDS byte addr
  u32x4_t g0;
  g0.x = 1u;                                               // count=1, user load descriptor
  g0.y = lds_addr;                                         // lds_addr [63:32]
  g0.z = (unsigned)(ga & 0xFFFFFFFFu);                     // global_addr [95:64]
  g0.w = (unsigned)((ga >> 32) & 0x1FFFFFFu) | (2u << 30); // global_addr hi + type=2 (image)
  i32x8_t g1;
  g1[0] = (1 << 16)        // data_size = 1 -> 2 bytes
        | (1 << 20)        // pad_enable
        | (4 << 22)        // pad_interval code 4 -> every 32 DWORDs (one 64-elem bf16 row)
        | (3 << 25);       // pad_amount  code 3 -> 4 DWORDs (8 bf16) => LDS ld = 72
  g1[1] = (int)((unsigned)Sdim << 16);  // tensor_dim0 = 2048 (lo16 in [31:16])
  g1[2] = (int)(64u << 16);             // tensor_dim0 hi = 0; tensor_dim1 = 64
  g1[3] = (int)(64u << 16);             // tensor_dim1 hi = 0; tile_dim0 = 64
  g1[4] = 64;                           // tile_dim1 = 64; tile_dim2 = 0
  g1[5] = Sdim;                         // tensor_dim0_stride[31:0] = 2048
  g1[6] = 0;                            // stride hi / tensor_dim1_stride lo
  g1[7] = 0;
  i32x4_t z4 = {0, 0, 0, 0};
  i32x8_t z8 = {0, 0, 0, 0, 0, 0, 0, 0};
  __builtin_amdgcn_tensor_load_to_lds(g0, g1, z4, z4, z8, 0);
}
#else
#define HAVE_TDM 0
#endif

// ---------------------------------------------------------------------------
// Kernel 1: Q/K/V projections.  C(4096x512) = X(4096x512) @ W^T + bias, bf16 out.
// ---------------------------------------------------------------------------
__global__ __launch_bounds__(128) void qkv_proj_kernel(
    const float* __restrict__ X,
    const float* __restrict__ Wq, const float* __restrict__ bq,
    const float* __restrict__ Wk, const float* __restrict__ bk,
    const float* __restrict__ Wv, const float* __restrict__ bv,
    bf16_t* __restrict__ Qb, bf16_t* __restrict__ Kb, bf16_t* __restrict__ Vb)
{
  __shared__ __align__(16) bf16_t At[64 * 72];
  __shared__ __align__(16) bf16_t Bt[64 * 72];
  const int tid = threadIdx.x, lane = tid & 31, wave = tid >> 5, half = lane >> 4;
  const int mt = blockIdx.x, nt = blockIdx.y, z = blockIdx.z;
  const float* W    = (z == 0) ? Wq : ((z == 1) ? Wk : Wv);
  const float* bias = (z == 0) ? bq : ((z == 1) ? bk : bv);
  bf16_t*      dst  = (z == 0) ? Qb : ((z == 1) ? Kb : Vb);

  v8f acc[4] = {};
  for (int kg = 0; kg < Ddim; kg += 64) {
    __syncthreads();
    for (int it = tid; it < 64 * 16; it += 128) {
      int r = it >> 4, c4 = (it & 15) << 2;
      float4 f = *(const float4*)&X[(size_t)(mt * 64 + r) * Ddim + kg + c4];
      bf16_t* d = &At[r * 72 + c4];
      d[0] = f2bf(f.x); d[1] = f2bf(f.y); d[2] = f2bf(f.z); d[3] = f2bf(f.w);
      float4 g = *(const float4*)&W[(size_t)(nt * 64 + r) * Ddim + kg + c4];
      bf16_t* e = &Bt[r * 72 + c4];
      e[0] = f2bf(g.x); e[1] = f2bf(g.y); e[2] = f2bf(g.z); e[3] = f2bf(g.w);
    }
    __syncthreads();
#pragma unroll
    for (int k0 = 0; k0 < 64; k0 += 32) {
      v16bf a = load_frag(At, 72, wave * 16, k0, lane);
#pragma unroll
      for (int f = 0; f < 4; ++f) {
        v16bf b = load_frag(Bt, 72, f * 16, k0, lane);
        acc[f] = wmma_bf16(a, b, acc[f]);
      }
    }
  }
#pragma unroll
  for (int f = 0; f < 4; ++f) {
    int col = nt * 64 + f * 16 + (lane & 15);
    float bb = bias[col];
#pragma unroll
    for (int e = 0; e < 8; ++e) {
      int row = mt * 64 + wave * 16 + e + half * 8;
      dst[(size_t)row * Hdim + col] = f2bf(acc[f][e] + bb);
    }
  }
}

// Vectorized transpose-gather of a 64x64 K tile: global b128 reads (8 bf16),
// strided LDS b16 stores into [col][k] layout.  Optionally prefetch next tile.
__device__ __forceinline__ void gather_k_tile(
    bf16_t* Kt, const bf16_t* __restrict__ Kb, size_t base, int st, int tid, bool pf)
{
  for (int it = tid; it < 64 * 8; it += 128) {
    int kk = it >> 3, n8 = (it & 7) << 3;
    uint4 v = *(const uint4*)&Kb[base + (size_t)kk * Sdim + st * 64 + n8];
    const bf16_t* s = (const bf16_t*)&v;
#pragma unroll
    for (int j = 0; j < 8; ++j) Kt[(n8 + j) * 72 + kk] = s[j];
    if (pf)
      __builtin_prefetch(&Kb[base + (size_t)kk * Sdim + (st + 1) * 64 + n8], 0, 3);
  }
}

// ---------------------------------------------------------------------------
// Kernel 2: online-softmax row statistics (max m, sum-exp l) of Q_h @ K_h.
// ---------------------------------------------------------------------------
__global__ __launch_bounds__(128) void attn_stats_kernel(
    const bf16_t* __restrict__ Qb, const bf16_t* __restrict__ Kb,
    float* __restrict__ Ms, float* __restrict__ Ls)
{
  __shared__ __align__(16) bf16_t Qt[64 * 72];
  __shared__ __align__(16) bf16_t Kt[64 * 72];
  const int tid = threadIdx.x, lane = tid & 31, wave = tid >> 5, half = lane >> 4;
  const int jt = blockIdx.x, bh = blockIdx.y;
  const size_t base = (size_t)bh * Sdim * DH;   // b*S*H + h*S*dh

  for (int it = tid; it < 64 * 16; it += 128) {
    int r = it >> 4, c4 = (it & 15) << 2;
    *(uint2*)&Qt[r * 72 + c4] = *(const uint2*)&Qb[base + (size_t)(jt * 64 + r) * DH + c4];
  }
  float m8[8], l8[8];
#pragma unroll
  for (int e = 0; e < 8; ++e) { m8[e] = -1e30f; l8[e] = 0.f; }

  for (int st = 0; st < Sdim / 64; ++st) {
    __syncthreads();
    gather_k_tile(Kt, Kb, base, st, tid, st + 1 < Sdim / 64);
    __syncthreads();
    v8f sc[4] = {};
#pragma unroll
    for (int k0 = 0; k0 < 64; k0 += 32) {
      v16bf a = load_frag(Qt, 72, wave * 16, k0, lane);
#pragma unroll
      for (int f = 0; f < 4; ++f) {
        v16bf b = load_frag(Kt, 72, f * 16, k0, lane);
        sc[f] = wmma_bf16(a, b, sc[f]);
      }
    }
#pragma unroll
    for (int e = 0; e < 8; ++e) {
      float tmax = fmaxf(fmaxf(sc[0][e], sc[1][e]), fmaxf(sc[2][e], sc[3][e]));
#pragma unroll
      for (int m = 1; m < 16; m <<= 1) tmax = fmaxf(tmax, __shfl_xor(tmax, m, 32));
      float mnew = fmaxf(m8[e], tmax);
      float s = __expf(sc[0][e] - mnew) + __expf(sc[1][e] - mnew) +
                __expf(sc[2][e] - mnew) + __expf(sc[3][e] - mnew);
#pragma unroll
      for (int m = 1; m < 16; m <<= 1) s += __shfl_xor(s, m, 32);
      l8[e] = l8[e] * __expf(m8[e] - mnew) + s;
      m8[e] = mnew;
    }
  }
  if ((lane & 15) == 0) {
#pragma unroll
    for (int e = 0; e < 8; ++e) {
      int row = jt * 64 + wave * 16 + e + half * 8;
      Ms[(size_t)bh * Sdim + row] = m8[e];
      Ls[(size_t)bh * Sdim + row] = l8[e];
    }
  }
}

// ---------------------------------------------------------------------------
// Kernel 3: out_h = V_h @ softmax(Q_h@K_h); recompute scores per j-block,
// normalize, re-stage as B operand through LDS, fuse V x W accumulation.
// V tile comes in via the Tensor Data Mover when available.
// ---------------------------------------------------------------------------
__global__ __launch_bounds__(128) void attn_out_kernel(
    const bf16_t* __restrict__ Qb, const bf16_t* __restrict__ Kb,
    const bf16_t* __restrict__ Vb,
    const float* __restrict__ Ms, const float* __restrict__ Ls,
    bf16_t* __restrict__ Ctx)
{
  __shared__ __align__(16) bf16_t Kt[64 * 72];
  __shared__ __align__(16) bf16_t Qt[64 * 72];
  __shared__ __align__(16) bf16_t Vt[64 * 72];
  __shared__ __align__(16) bf16_t Wt[64 * 72];
  const int tid = threadIdx.x, lane = tid & 31, wave = tid >> 5, half = lane >> 4;
  const int st = blockIdx.x, bh = blockIdx.y;
  const int b = bh >> 3, h = bh & 7;
  const size_t base = (size_t)bh * Sdim * DH;

  gather_k_tile(Kt, Kb, base, st, tid, false);   // K tile fixed for this column block

  v8f o[4] = {};
  for (int jb = 0; jb < Sdim / 64; ++jb) {
    __syncthreads();
#if HAVE_TDM
    if (wave == 0) tdm_load_64x64_pad72(&Vb[base + jb * 64], Vt);  // V 64x64, stride Sdim
    for (int it = tid; it < 64 * 16; it += 128) {
      int r = it >> 4, c4 = (it & 15) << 2;
      *(uint2*)&Qt[r * 72 + c4] = *(const uint2*)&Qb[base + (size_t)(jb * 64 + r) * DH + c4];
    }
    if (wave == 0) __builtin_amdgcn_s_wait_tensorcnt(0);
#else
    for (int it = tid; it < 64 * 16; it += 128) {
      int r = it >> 4, c4 = (it & 15) << 2;
      *(uint2*)&Qt[r * 72 + c4] = *(const uint2*)&Qb[base + (size_t)(jb * 64 + r) * DH + c4];
      *(uint2*)&Vt[r * 72 + c4] = *(const uint2*)&Vb[base + (size_t)r * Sdim + jb * 64 + c4];
    }
#endif
    __syncthreads();
    v8f sc[4] = {};
#pragma unroll
    for (int k0 = 0; k0 < 64; k0 += 32) {
      v16bf a = load_frag(Qt, 72, wave * 16, k0, lane);
#pragma unroll
      for (int f = 0; f < 4; ++f) {
        v16bf bb = load_frag(Kt, 72, f * 16, k0, lane);
        sc[f] = wmma_bf16(a, bb, sc[f]);
      }
    }
#pragma unroll
    for (int e = 0; e < 8; ++e) {
      int jrow = jb * 64 + wave * 16 + e + half * 8;
      float m    = Ms[(size_t)bh * Sdim + jrow];
      float invl = 1.0f / Ls[(size_t)bh * Sdim + jrow];
#pragma unroll
      for (int f = 0; f < 4; ++f) {
        float w = __expf(sc[f][e] - m) * invl;
        Wt[(f * 16 + (lane & 15)) * 72 + (wave * 16 + e + half * 8)] = f2bf(w);
      }
    }
    __syncthreads();
#pragma unroll
    for (int k0 = 0; k0 < 64; k0 += 32) {
      v16bf a = load_frag(Vt, 72, wave * 16, k0, lane);
#pragma unroll
      for (int f = 0; f < 4; ++f) {
        v16bf bb = load_frag(Wt, 72, f * 16, k0, lane);
        o[f] = wmma_bf16(a, bb, o[f]);
      }
    }
  }
#pragma unroll
  for (int f = 0; f < 4; ++f) {
    int s2 = st * 64 + f * 16 + (lane & 15);
    int token = b * Sdim + s2;
#pragma unroll
    for (int e = 0; e < 8; ++e) {
      int d = wave * 16 + e + half * 8;
      Ctx[(size_t)token * Hdim + h * DH + d] = f2bf(o[f][e]);
    }
  }
}

// ---------------------------------------------------------------------------
// Kernel 4: Y(4096x512, fp32) = ctx(bf16) @ out_w^T + out_b
// ---------------------------------------------------------------------------
__global__ __launch_bounds__(128) void out_proj_kernel(
    const bf16_t* __restrict__ Ctx, const float* __restrict__ Wo,
    const float* __restrict__ bo, float* __restrict__ Y)
{
  __shared__ __align__(16) bf16_t At[64 * 72];
  __shared__ __align__(16) bf16_t Bt[64 * 72];
  const int tid = threadIdx.x, lane = tid & 31, wave = tid >> 5, half = lane >> 4;
  const int mt = blockIdx.x, nt = blockIdx.y;

  v8f acc[4] = {};
  for (int kg = 0; kg < Hdim; kg += 64) {
    __syncthreads();
    for (int it = tid; it < 64 * 16; it += 128) {
      int r = it >> 4, c4 = (it & 15) << 2;
      *(uint2*)&At[r * 72 + c4] = *(const uint2*)&Ctx[(size_t)(mt * 64 + r) * Hdim + kg + c4];
      float4 g = *(const float4*)&Wo[(size_t)(nt * 64 + r) * Hdim + kg + c4];
      bf16_t* e2 = &Bt[r * 72 + c4];
      e2[0] = f2bf(g.x); e2[1] = f2bf(g.y); e2[2] = f2bf(g.z); e2[3] = f2bf(g.w);
    }
    __syncthreads();
#pragma unroll
    for (int k0 = 0; k0 < 64; k0 += 32) {
      v16bf a = load_frag(At, 72, wave * 16, k0, lane);
#pragma unroll
      for (int f = 0; f < 4; ++f) {
        v16bf b = load_frag(Bt, 72, f * 16, k0, lane);
        acc[f] = wmma_bf16(a, b, acc[f]);
      }
    }
  }
#pragma unroll
  for (int f = 0; f < 4; ++f) {
    int col = nt * 64 + f * 16 + (lane & 15);
    float bb = bo[col];
#pragma unroll
    for (int e = 0; e < 8; ++e) {
      int row = mt * 64 + wave * 16 + e + half * 8;
      Y[(size_t)row * Ddim + col] = acc[f][e] + bb;
    }
  }
}

extern "C" void kernel_launch(void* const* d_in, const int* in_sizes, int n_in,
                              void* d_out, int out_size, void* d_ws, size_t ws_size,
                              hipStream_t stream) {
  const float* data = (const float*)d_in[0];
  const float* Wq = (const float*)d_in[1];
  const float* bq = (const float*)d_in[2];
  const float* Wk = (const float*)d_in[3];
  const float* bk = (const float*)d_in[4];
  const float* Wv = (const float*)d_in[5];
  const float* bv = (const float*)d_in[6];
  const float* Wo = (const float*)d_in[7];
  const float* bo = (const float*)d_in[8];
  float* Y = (float*)d_out;

  const size_t NB = (size_t)NTOK * Hdim;            // 2,097,152 elements
  char* ws = (char*)d_ws;
  bf16_t* Qb  = (bf16_t*)(ws + 0 * NB * 2);
  bf16_t* Kb  = (bf16_t*)(ws + 1 * NB * 2);
  bf16_t* Vb  = (bf16_t*)(ws + 2 * NB * 2);
  bf16_t* Ctx = (bf16_t*)(ws + 3 * NB * 2);
  float*  Ms  = (float*) (ws + 4 * NB * 2);
  float*  Ls  = Ms + (size_t)(Bsz * NHEAD) * Sdim;

  dim3 blk(128);
  qkv_proj_kernel<<<dim3(NTOK / 64, Hdim / 64, 3), blk, 0, stream>>>(
      data, Wq, bq, Wk, bk, Wv, bv, Qb, Kb, Vb);
  attn_stats_kernel<<<dim3(Sdim / 64, Bsz * NHEAD), blk, 0, stream>>>(Qb, Kb, Ms, Ls);
  attn_out_kernel<<<dim3(Sdim / 64, Bsz * NHEAD), blk, 0, stream>>>(Qb, Kb, Vb, Ms, Ls, Ctx);
  out_proj_kernel<<<dim3(NTOK / 64, Hdim / 64), blk, 0, stream>>>(Ctx, Wo, bo, Y);
}
